// DynamicGraphAttention_24043226923413
// MI455X (gfx1250) — matrix operand order repaired and checked by reading, not compiled, not run on hardware
//
#include <hip/hip_runtime.h>
#include <hip/hip_bf16.h>
#include <math.h>

typedef __attribute__((ext_vector_type(16))) _Float16 v16h;
typedef __attribute__((ext_vector_type(8)))  float    v8f;
typedef __attribute__((ext_vector_type(4)))  unsigned int u32x4;
typedef __attribute__((ext_vector_type(8)))  int      i32x8;
typedef __attribute__((ext_vector_type(4)))  int      i32x4;

#define N_NODES 2048
#define DMODEL  512
#define NHEADS  8
#define DK      64
#define NEG_BIG (-1.0e30f)

#define KTILE_LDS 2560   // 32 keys x 80 halves (64 data + 16 pad)

// ---------------------------------------------------------------------------
// WMMA helpers (CDNA5 16x16x32 f16 -> f32)
// ---------------------------------------------------------------------------
__device__ __forceinline__ v8f wmma_f16(v16h a, v16h b, v8f c) {
  return __builtin_amdgcn_wmma_f32_16x16x32_f16(
      /*neg_a=*/false, a, /*neg_b=*/false, b,
      /*c_mod=*/(short)0, c, /*reuse_a=*/false, /*reuse_b=*/false);
}

// A fragment, 16x32 (MxK), 16-bit elements. Per ISA 7.12.2:
// lanes 0-15: M=lane, K = {0..7} in v0..3, {16..23} in v4..7
// lanes16-31: M=lane-16, K = {8..15} in v0..3, {24..31} in v4..7
__device__ __forceinline__ v16h load_a_f32(const float* base, int ld, int m0, int k0) {
  int lane = threadIdx.x & 31;
  int row  = m0 + (lane & 15);
  int koff = (lane >> 4) * 8;
  const float* p0 = base + (size_t)row * ld + k0 + koff;
  const float* p1 = p0 + 16;
  v16h a;
#pragma unroll
  for (int i = 0; i < 8; ++i) { a[i] = (_Float16)p0[i]; a[i + 8] = (_Float16)p1[i]; }
  return a;
}

__device__ __forceinline__ v16h load_a_f16(const _Float16* base, int ld, int m0, int k0) {
  int lane = threadIdx.x & 31;
  int row  = m0 + (lane & 15);
  int koff = (lane >> 4) * 8;
  const _Float16* p0 = base + (size_t)row * ld + k0 + koff;
  const _Float16* p1 = p0 + 16;
  v16h a;
#pragma unroll
  for (int i = 0; i < 8; ++i) { a[i] = p0[i]; a[i + 8] = p1[i]; }
  return a;
}

// B fragment, 32x16 (KxN), with B[k][n] = base[(n0+n)*ld + k0 + k]
// lane = N, lane-half h: K = 8h..8h+7 in elems 0..7, 16+8h..16+8h+7 in elems 8..15.
__device__ __forceinline__ v16h load_b_f32(const float* base, int ld, int n0, int k0) {
  int lane = threadIdx.x & 31;
  int row  = n0 + (lane & 15);
  int koff = (lane >> 4) * 8;
  const float* p = base + (size_t)row * ld + k0 + koff;
  v16h b;
#pragma unroll
  for (int i = 0; i < 8; ++i) { b[i] = (_Float16)p[i]; b[i + 8] = (_Float16)p[i + 16]; }
  return b;
}

__device__ __forceinline__ v16h load_b_f16(const _Float16* base, int ld, int n0, int k0) {
  int lane = threadIdx.x & 31;
  int row  = n0 + (lane & 15);
  int koff = (lane >> 4) * 8;
  const _Float16* p = base + (size_t)row * ld + k0 + koff;
  v16h b;
#pragma unroll
  for (int i = 0; i < 8; ++i) { b[i] = p[i]; b[i + 8] = p[i + 16]; }
  return b;
}

// ---------------------------------------------------------------------------
// TDM: async DMA of one K tile (32 keys x 64 halves) into LDS with row padding
// to stride 80 halves (tile_dim0 = 32 DWORDs -> pad_interval=4, pad 8 DWORDs
// -> pad_amount=7). Descriptor packing per CDNA5 ISA §8.3/8.4.
// Toolchain here uses the 6-arg builtin (clang-23 / therock-10.0).
// ---------------------------------------------------------------------------
#if __has_builtin(__builtin_amdgcn_tensor_load_to_lds)
#define HAVE_TDM 1
__device__ __forceinline__ void tdm_load_ktile(const _Float16* gsrc, _Float16* lds_dst) {
  unsigned lds_off = (unsigned)(size_t)lds_dst;                 // low 32b = LDS offset
  unsigned long long ga = (unsigned long long)(size_t)gsrc;
  u32x4 g0;
  g0[0] = 1u;                                                   // count=1
  g0[1] = lds_off;                                              // lds_addr
  g0[2] = (unsigned)(ga & 0xffffffffu);                         // global_addr[31:0]
  g0[3] = (unsigned)((ga >> 32) & 0x01ffffffu) | 0x80000000u;   // addr[56:32] | type=2
  i32x8 g1;
  g1[0] = (int)((1u << 16)          // data_size = 2 bytes
              | (1u << 20)          // pad_enable
              | (4u << 22)          // pad_interval: 32 DWORDs (=128B row)
              | (7u << 25));        // pad_amount: 8 DWORDs (=32B pad)
  g1[1] = (int)(64u << 16);         // tensor_dim0 = 64 (bits 63:48)
  g1[2] = (int)(32u << 16);         // tensor_dim1 = 32 (bits 95:80)
  g1[3] = (int)(64u << 16);         // tile_dim0 = 64 (bits 127:112)
  g1[4] = 32;                       // tile_dim1 = 32 (bits 143:128)
  g1[5] = 512;                      // tensor_dim0_stride = 512 elems (bits 191:160)
  g1[6] = 0;
  g1[7] = 0;
  i32x4 z4 = {0, 0, 0, 0};
  i32x8 z8 = {0, 0, 0, 0, 0, 0, 0, 0};
  __builtin_amdgcn_tensor_load_to_lds(g0, g1, z4, z4, z8, 0);
}
#else
#define HAVE_TDM 0
#endif

// ---------------------------------------------------------------------------
// Kernel 1: fused QKV projection  Y = X @ W^T + b  -> f16 (4096 x 512) each
// grid (64, 8, 3), block 128 (4 waves); wave = 16 rows x 64 cols of output
// ---------------------------------------------------------------------------
__global__ void qkv_proj_kernel(const float* __restrict__ x,
                                const float* __restrict__ wq, const float* __restrict__ bq,
                                const float* __restrict__ wk, const float* __restrict__ bk,
                                const float* __restrict__ wv, const float* __restrict__ bv,
                                _Float16* __restrict__ qout, _Float16* __restrict__ kout,
                                _Float16* __restrict__ vout) {
  const float* w; const float* bias; _Float16* out;
  if (blockIdx.z == 0)      { w = wq; bias = bq; out = qout; }
  else if (blockIdx.z == 1) { w = wk; bias = bk; out = kout; }
  else                      { w = wv; bias = bv; out = vout; }

  int wave = threadIdx.x >> 5, lane = threadIdx.x & 31;
  int m0 = blockIdx.x * 64 + wave * 16;
  int n0 = blockIdx.y * 64;

  v8f zero = {0.f, 0.f, 0.f, 0.f, 0.f, 0.f, 0.f, 0.f};
  v8f acc[4] = {zero, zero, zero, zero};

  for (int k0 = 0; k0 < DMODEL; k0 += 32) {
    v16h a = load_a_f32(x, DMODEL, m0, k0);
#pragma unroll
    for (int t = 0; t < 4; ++t) {
      v16h bf = load_b_f32(w, DMODEL, n0 + t * 16, k0);   // B[k][n] = W[n][k]
      acc[t] = wmma_f16(a, bf, acc[t]);
    }
  }

  int half = lane >> 4, col = lane & 15;
#pragma unroll
  for (int t = 0; t < 4; ++t) {
    int n = n0 + t * 16 + col;
    float bn = bias[n];
#pragma unroll
    for (int r = 0; r < 8; ++r) {
      int m = m0 + half * 8 + r;
      out[(size_t)m * DMODEL + n] = (_Float16)(acc[t][r] + bn);
    }
  }
}

// ---------------------------------------------------------------------------
// Kernel 2: flash attention with graph mask + temporal bias
// grid (N/128, B*H), block 256 (8 waves). Wave owns 16 query rows.
// K tile double-buffered via TDM async DMA; V staged transposed manually.
// ---------------------------------------------------------------------------
__global__ void __launch_bounds__(256)
attn_kernel(const _Float16* __restrict__ qb, const _Float16* __restrict__ kb,
            const _Float16* __restrict__ vb, const int* __restrict__ adj,
            const float* __restrict__ et, const int* __restrict__ ctp,
            const float* __restrict__ wt, const float* __restrict__ bt,
            _Float16* __restrict__ att) {
  __shared__ __align__(16) _Float16 smem[2 * KTILE_LDS + 64 * 40 + 8 * 16 * 40];
  // smem + cur*KTILE_LDS : K tile buffers (32 keys x 64 dims, ld 80), double buffered
  _Float16* Vt = smem + 2 * KTILE_LDS;            // 64 dims x 32 keys, ld 40 (V transposed)

  int tid = threadIdx.x;
  int wave = tid >> 5, lane = tid & 31;
  _Float16* Sw = smem + 2 * KTILE_LDS + 64 * 40 + wave * (16 * 40);  // 16x32 P scratch

  int bh = blockIdx.y;
  int b = bh >> 3, h = bh & 7;
  int q0 = blockIdx.x * 128 + wave * 16;

  float ct  = (float)(*ctp);
  float wth = wt[h];
  float bth = bt[h];

  const _Float16* qhb = qb + (size_t)b * N_NODES * DMODEL + h * DK;
  const _Float16* khb = kb + (size_t)b * N_NODES * DMODEL + h * DK;
  const _Float16* vhb = vb + (size_t)b * N_NODES * DMODEL + h * DK;
  const int*   adjb = adj + (size_t)b * N_NODES * N_NODES;
  const float* etb  = et  + (size_t)b * N_NODES * N_NODES;

  // Q fragments for this wave's 16 rows (dk = 64 -> two K=32 chunks)
  v16h aq0 = load_a_f16(qhb, DMODEL, q0, 0);
  v16h aq1 = load_a_f16(qhb, DMODEL, q0, 32);

  v8f zero = {0.f, 0.f, 0.f, 0.f, 0.f, 0.f, 0.f, 0.f};
  v8f o[4] = {zero, zero, zero, zero};
  float mrow[8], lrow[8];
#pragma unroll
  for (int r = 0; r < 8; ++r) { mrow[r] = NEG_BIG; lrow[r] = 0.f; }

  int half = lane >> 4, col = lane & 15;
  int cur = 0;

#if HAVE_TDM
  if (wave == 0) tdm_load_ktile(khb, smem);       // prologue: async K tile 0 -> buf0
#endif

  for (int kt = 0; kt < N_NODES; kt += 32) {
    _Float16* Kcur = smem + cur * KTILE_LDS;
#if HAVE_TDM
    __builtin_amdgcn_s_wait_tensorcnt(0);         // K tile `kt` landed (no-op for waves 1..7)
#endif
    __syncthreads();                              // publish K tile; prev compute done

#if !HAVE_TDM
    // fallback: manual K staging (row-major, 8 contiguous halves per thread)
    {
      int e = tid * 8;
      int key = e >> 6, d = e & 63;
      const _Float16* src = khb + (size_t)(kt + key) * DMODEL + d;
      _Float16* dst = Kcur + key * 80 + d;
#pragma unroll
      for (int i = 0; i < 8; ++i) dst[i] = src[i];
    }
#endif
    // stage V tile transposed: Vt[d][key] = V[key][d]
#pragma unroll
    for (int i = 0; i < 8; ++i) {
      int e = tid + 256 * i;
      int key = e >> 6, d = e & 63;
      Vt[d * 40 + key] = vhb[(size_t)(kt + key) * DMODEL + d];
    }
#if HAVE_TDM
    // kick off next K tile DMA; overlaps with compute below
    if (wave == 0 && kt + 32 < N_NODES)
      tdm_load_ktile(khb + (size_t)(kt + 32) * DMODEL, smem + (cur ^ 1) * KTILE_LDS);
#endif
    if (kt + 32 < N_NODES)
      __builtin_prefetch(vhb + (size_t)(kt + 32 + (tid & 31)) * DMODEL, 0, 1);
    __syncthreads();                              // Vt ready for all waves

    // S = (Q K^T) * 1/sqrt(dk): two 16x16 column tiles over the 32 keys
    v8f s0 = zero, s1 = zero;
    s0 = wmma_f16(aq0, load_b_f16(Kcur, 80, 0, 0),  s0);
    s0 = wmma_f16(aq1, load_b_f16(Kcur, 80, 0, 32), s0);
    s1 = wmma_f16(aq0, load_b_f16(Kcur, 80, 16, 0),  s1);
    s1 = wmma_f16(aq1, load_b_f16(Kcur, 80, 16, 32), s1);

    // mask + temporal bias epilogue (per-element gathers; arrays are L2-resident)
    float p0[8], p1[8];
#pragma unroll
    for (int r = 0; r < 8; ++r) {
      int qi = q0 + half * 8 + r;
      size_t rowoff = (size_t)qi * N_NODES;
      int j0 = kt + col, j1 = kt + 16 + col;
      float s0v = s0[r] * 0.125f;
      float s1v = s1[r] * 0.125f;
      if (adjb[rowoff + j0] == 0) s0v = NEG_BIG;
      if (adjb[rowoff + j1] == 0) s1v = NEG_BIG;
      float dt0 = ct - etb[rowoff + j0]; dt0 = dt0 > 0.f ? dt0 : 0.f;
      float dt1 = ct - etb[rowoff + j1]; dt1 = dt1 > 0.f ? dt1 : 0.f;
      s0v += __expf(-0.1f * dt0) * wth + bth;
      s1v += __expf(-0.1f * dt1) * wth + bth;
      p0[r] = s0v; p1[r] = s1v;
    }

    // online softmax: row reductions across the 16 lanes of each half-wave
#pragma unroll
    for (int r = 0; r < 8; ++r) {
      float mx = fmaxf(p0[r], p1[r]);
#pragma unroll
      for (int msk = 8; msk >= 1; msk >>= 1) mx = fmaxf(mx, __shfl_xor(mx, msk, 32));
      float mnew  = fmaxf(mrow[r], mx);
      float alpha = __expf(mrow[r] - mnew);
      mrow[r] = mnew;
      float pe0 = __expf(p0[r] - mnew);
      float pe1 = __expf(p1[r] - mnew);
      float rs = pe0 + pe1;
#pragma unroll
      for (int msk = 8; msk >= 1; msk >>= 1) rs += __shfl_xor(rs, msk, 32);
      lrow[r] = lrow[r] * alpha + rs;
#pragma unroll
      for (int t = 0; t < 4; ++t) o[t][r] *= alpha;
      int M = half * 8 + r;
      Sw[M * 40 + col]      = (_Float16)pe0;
      Sw[M * 40 + 16 + col] = (_Float16)pe1;
    }

    // O += P V : one A-frag (16x32 over keys), four 16-wide dk tiles
    v16h ap = load_a_f16(Sw, 40, 0, 0);
#pragma unroll
    for (int t = 0; t < 4; ++t)
      o[t] = wmma_f16(ap, load_b_f16(Vt, 40, t * 16, 0), o[t]);

    cur ^= 1;
  }

  // normalize and store into (B*N, 512) f16 buffer at head slice
#pragma unroll
  for (int t = 0; t < 4; ++t) {
#pragma unroll
    for (int r = 0; r < 8; ++r) {
      int qi = q0 + half * 8 + r;
      int dc = h * DK + t * 16 + col;
      att[(size_t)(b * N_NODES + qi) * DMODEL + dc] = (_Float16)(o[t][r] / lrow[r]);
    }
  }
}

// ---------------------------------------------------------------------------
// Kernel 3: output projection + bias + residual -> f32
// grid (64, 8), block 128
// ---------------------------------------------------------------------------
__global__ void out_proj_kernel(const _Float16* __restrict__ att,
                                const float* __restrict__ wo, const float* __restrict__ bo,
                                const float* __restrict__ x, float* __restrict__ out) {
  int wave = threadIdx.x >> 5, lane = threadIdx.x & 31;
  int m0 = blockIdx.x * 64 + wave * 16;
  int n0 = blockIdx.y * 64;

  v8f zero = {0.f, 0.f, 0.f, 0.f, 0.f, 0.f, 0.f, 0.f};
  v8f acc[4] = {zero, zero, zero, zero};

  for (int k0 = 0; k0 < DMODEL; k0 += 32) {
    v16h a = load_a_f16(att, DMODEL, m0, k0);
#pragma unroll
    for (int t = 0; t < 4; ++t) {
      v16h bf = load_b_f32(wo, DMODEL, n0 + t * 16, k0);
      acc[t] = wmma_f16(a, bf, acc[t]);
    }
  }

  int half = lane >> 4, col = lane & 15;
#pragma unroll
  for (int t = 0; t < 4; ++t) {
    int n = n0 + t * 16 + col;
    float bn = bo[n];
#pragma unroll
    for (int r = 0; r < 8; ++r) {
      int m = m0 + half * 8 + r;
      out[(size_t)m * DMODEL + n] = acc[t][r] + bn + x[(size_t)m * DMODEL + n];
    }
  }
}

// ---------------------------------------------------------------------------
extern "C" void kernel_launch(void* const* d_in, const int* in_sizes, int n_in,
                              void* d_out, int out_size, void* d_ws, size_t ws_size,
                              hipStream_t stream) {
  const float* x   = (const float*)d_in[0];
  const int*   adj = (const int*)d_in[1];
  const float* et  = (const float*)d_in[2];
  const int*   ctp = (const int*)d_in[3];
  const float* wq  = (const float*)d_in[4];
  const float* bq  = (const float*)d_in[5];
  const float* wk  = (const float*)d_in[6];
  const float* bk  = (const float*)d_in[7];
  const float* wv  = (const float*)d_in[8];
  const float* bv  = (const float*)d_in[9];
  const float* wo  = (const float*)d_in[10];
  const float* bo  = (const float*)d_in[11];
  const float* wt  = (const float*)d_in[12];
  const float* bt  = (const float*)d_in[13];
  float* out = (float*)d_out;

  const size_t MAT = (size_t)2 * N_NODES * DMODEL;  // 4096*512 halves
  _Float16* ws   = (_Float16*)d_ws;
  _Float16* qbuf = ws;
  _Float16* kbuf = ws + MAT;
  _Float16* vbuf = ws + 2 * MAT;
  _Float16* attb = ws + 3 * MAT;

  qkv_proj_kernel<<<dim3(64, 8, 3), 128, 0, stream>>>(x, wq, bq, wk, bk, wv, bv,
                                                      qbuf, kbuf, vbuf);
  attn_kernel<<<dim3(N_NODES / 128, 2 * NHEADS), 256, 0, stream>>>(
      qbuf, kbuf, vbuf, adj, et, ctp, wt, bt, attb);
  out_proj_kernel<<<dim3(64, 8), 128, 0, stream>>>(attb, wo, bo, x, out);
}
